// ContextNodeEdgeNet_14035953123600
// MI455X (gfx1250) — matrix operand order repaired and checked by reading, not compiled, not run on hardware
//
#include <hip/hip_runtime.h>
#include <hip/hip_bf16.h>

// ---------------------------------------------------------------------------
// ContextNodeEdgeNet on MI455X (gfx1250): bf16 WMMA GEMM pipeline, v4
//   - 128x64 block tile, 8 wave32s, 32x32 per wave -> 4 v_wmma per K-step
//   - chunk-uniform staging, zero-fill for OOB chunks (requires K%8==0, N%4==0,
//     which holds for every GEMM in this network; M may be ragged)
//   - fp32->bf16 via hardware cvt (fptrunc pair -> v_cvt_pk_bf16_f32)
// ---------------------------------------------------------------------------

typedef __attribute__((ext_vector_type(16))) __bf16 v16bf;
typedef __attribute__((ext_vector_type(2)))  __bf16 v2bf;
typedef __attribute__((ext_vector_type(8)))  float  v8f;

#define NNODE 10000
#define NEDGE 160000
#define LDSPAD 40   // ushorts per LDS row (32 data + 8 pad), 80B = 5 x 16B

__device__ __forceinline__ unsigned pack2bf(float lo, float hi) {
    v2bf p;
    p.x = (__bf16)lo;       // RNE fptrunc -> hardware cvt
    p.y = (__bf16)hi;
    return __builtin_bit_cast(unsigned, p);
}

// C[M,N] = epilogue( A[g(M),K] @ W[wrow0:wrow0+K, 0:N] + bias )
//   gth     : optional row-gather index for A (length M)
//   beta=1  : accumulate onto existing C (epi==0 only)
//   epi==1  : C *= sigmoid(acc + bias)   (fused gating)
// Constraints (satisfied by all call sites): K % 8 == 0, N % 4 == 0.
__global__ __launch_bounds__(256) void gemm_bf16_wmma(
    const float* __restrict__ A, int lda, const int* __restrict__ gth,
    const float* __restrict__ W, int ldw, int wrow0,
    const float* __restrict__ bias,
    float* __restrict__ C, int ldc,
    int M, int N, int K, int beta, int epi)
{
    __shared__ unsigned short Al[128 * LDSPAD];
    __shared__ unsigned short Bl[64 * LDSPAD];
    const int tid  = threadIdx.x;
    const int lane = tid & 31;
    const int wv   = tid >> 5;
    const int wm   = wv & 3;            // 4 M-subtiles of 32
    const int wn   = wv >> 2;           // 2 N-halves of 32
    const int m0   = blockIdx.y << 7;   // BM = 128
    const int n0   = blockIdx.x << 6;   // BN = 64

    v8f acc[4];                         // [mi*2+ni], mi,ni in {0,1}
    #pragma unroll
    for (int i = 0; i < 4; ++i) acc[i] = (v8f){0.f,0.f,0.f,0.f,0.f,0.f,0.f,0.f};

    // ---- A staging assignment: 2 chunks of (row, 8 consecutive K) ----------
    const int ar = tid >> 2;            // 0..63
    const int ak = (tid & 3) << 3;      // 0/8/16/24
    bool aok[2]; long long abase[2];
    #pragma unroll
    for (int cch = 0; cch < 2; ++cch) {
        const int row = m0 + ar + (cch << 6);
        aok[cch] = (row < M);
        long long g = 0;
        if (aok[cch]) g = gth ? gth[row] : row;
        abase[cch] = g * lda;
    }
    // ---- B staging assignment: (k,k+1) pair x 4 columns --------------------
    const int bk  = (tid >> 4) << 1;    // 0,2,..,30
    const int bn4 = (tid & 15) << 2;    // 0,4,..,60
    const bool bfull = (n0 + bn4 + 4 <= N);   // N%4==0 -> chunk all-in or all-out

    for (int k0 = 0; k0 < K; k0 += 32) {
        // stage A -> Al[row][k] (bf16, packed b128 store); K%8==0 -> chunk all-in/all-out
        #pragma unroll
        for (int cch = 0; cch < 2; ++cch) {
            const int kbeg = k0 + ak;
            uint4 pk = {0u, 0u, 0u, 0u};
            if (aok[cch] && kbeg + 8 <= K) {
                const float* ap = A + abase[cch] + kbeg;
                const float4 f0 = *(const float4*)ap;
                const float4 f1 = *(const float4*)(ap + 4);
                pk.x = pack2bf(f0.x, f0.y); pk.y = pack2bf(f0.z, f0.w);
                pk.z = pack2bf(f1.x, f1.y); pk.w = pack2bf(f1.z, f1.w);
            }
            *(uint4*)&Al[(ar + (cch << 6)) * LDSPAD + ak] = pk;
        }
        // stage B (transposed) -> Bl[n][k]; K%2==0 -> k-pair all-in/all-out
        {
            const int kg = k0 + bk;
            unsigned pw[4] = {0u, 0u, 0u, 0u};
            if (bfull && kg + 2 <= K) {
                const float4 q0 = *(const float4*)(W + (long long)(wrow0 + kg    ) * ldw + n0 + bn4);
                const float4 q1 = *(const float4*)(W + (long long)(wrow0 + kg + 1) * ldw + n0 + bn4);
                pw[0] = pack2bf(q0.x, q1.x);
                pw[1] = pack2bf(q0.y, q1.y);
                pw[2] = pack2bf(q0.z, q1.z);
                pw[3] = pack2bf(q0.w, q1.w);
            }
            #pragma unroll
            for (int j = 0; j < 4; ++j)
                *(unsigned*)&Bl[(bn4 + j) * LDSPAD + bk] = pw[j];
        }
        __syncthreads();

        const int lh = lane >> 4;
        const int ll = lane & 15;
        union Frag { v16bf v; uint4 q[2]; } fa0, fa1, fb0, fb1;
        // A 16x32 bf16 frag: elems 0..7 -> K=8*lh+0..7 ; 8..15 -> K=16+8*lh+0..7
        {
            const uint4* p0 = (const uint4*)&Al[(wm * 32 +      ll) * LDSPAD + (lh << 3)];
            fa0.q[0] = p0[0]; fa0.q[1] = p0[2];
            const uint4* p1 = (const uint4*)&Al[(wm * 32 + 16 + ll) * LDSPAD + (lh << 3)];
            fa1.q[0] = p1[0]; fa1.q[1] = p1[2];
        }
        // B 32x16 bf16 frag: elems 0..7 -> K=8*lh+0..7 ; 8..15 -> K=16+8*lh+0..7
        {
            const uint4* p0 = (const uint4*)&Bl[(wn * 32 +      ll) * LDSPAD + (lh << 3)];
            fb0.q[0] = p0[0]; fb0.q[1] = p0[2];
            const uint4* p1 = (const uint4*)&Bl[(wn * 32 + 16 + ll) * LDSPAD + (lh << 3)];
            fb1.q[0] = p1[0]; fb1.q[1] = p1[2];
        }
        acc[0] = __builtin_amdgcn_wmma_f32_16x16x32_bf16(false, fa0.v, false, fb0.v, (short)0, acc[0], false, false);
        acc[1] = __builtin_amdgcn_wmma_f32_16x16x32_bf16(false, fa0.v, false, fb1.v, (short)0, acc[1], false, false);
        acc[2] = __builtin_amdgcn_wmma_f32_16x16x32_bf16(false, fa1.v, false, fb0.v, (short)0, acc[2], false, false);
        acc[3] = __builtin_amdgcn_wmma_f32_16x16x32_bf16(false, fa1.v, false, fb1.v, (short)0, acc[3], false, false);
        __syncthreads();
    }

    const int lh = lane >> 4, ll = lane & 15;
    #pragma unroll
    for (int ni = 0; ni < 2; ++ni) {
        const int cc = n0 + wn * 32 + ni * 16 + ll;
        if (cc >= N) continue;
        const float bb = bias ? bias[cc] : 0.f;
        #pragma unroll
        for (int mi = 0; mi < 2; ++mi) {
            const v8f a = acc[mi * 2 + ni];
            #pragma unroll
            for (int v = 0; v < 8; ++v) {
                const int r = m0 + wm * 32 + mi * 16 + v + (lh << 3);  // M = v + 8*(lane>>4)
                if (r >= M) continue;
                const long long idx = (long long)r * ldc + cc;
                const float s = a[v] + bb;
                if (epi) C[idx] *= 1.f / (1.f + __expf(-s));
                else     C[idx] = s + (beta ? C[idx] : 0.f);
            }
        }
    }
}

// One block per row; blockDim == D (32/128/256). Optional residual R, optional ReLU.
__global__ void ln_kernel(const float* __restrict__ X, const float* __restrict__ R,
                          const float* __restrict__ g, const float* __restrict__ b,
                          float* __restrict__ Y, int D, int relu)
{
    __shared__ float red[256];
    const int row = blockIdx.x;
    const int t = threadIdx.x;
    const long long base = (long long)row * D;
    float x = X[base + t];
    if (R) x += R[base + t];
    red[t] = x; __syncthreads();
    for (int s = blockDim.x >> 1; s > 0; s >>= 1) {
        if (t < s) red[t] += red[t + s];
        __syncthreads();
    }
    const float mu = red[0] / D;
    __syncthreads();
    const float d = x - mu;
    red[t] = d * d; __syncthreads();
    for (int s = blockDim.x >> 1; s > 0; s >>= 1) {
        if (t < s) red[t] += red[t + s];
        __syncthreads();
    }
    const float var = red[0] / D;
    float y = d * rsqrtf(var + 1e-5f) * g[t] + b[t];
    if (relu) y = fmaxf(y, 0.f);
    Y[base + t] = y;
}

// out[e] = e_feat[e] + nodes[i0[e]] + nodes[i1[e]]   (blockDim == D)
__global__ void add3_gather(const float* __restrict__ ef, const float* __restrict__ nod,
                            const int* __restrict__ i0, const int* __restrict__ i1,
                            float* __restrict__ out, int D)
{
    const long long r = blockIdx.x; const int t = threadIdx.x;
    const int a = i0[r], b = i1[r];
    out[r * D + t] = ef[r * D + t] + nod[(long long)a * D + t] + nod[(long long)b * D + t];
}

// dst[idx[e]] += src[e]   (segment_sum; blockDim == D)
__global__ void scatter_add(const float* __restrict__ src, const int* __restrict__ idx,
                            float* __restrict__ dst, int D)
{
    const long long r = blockIdx.x; const int t = threadIdx.x;
    atomicAdd(&dst[(long long)idx[r] * D + t], src[r * D + t]);
}

__global__ void fill_zero(float* __restrict__ p, long long n)
{
    const long long i = (long long)blockIdx.x * blockDim.x + threadIdx.x;
    if (i < n) p[i] = 0.f;
}

__global__ void add_inplace(float* __restrict__ a, const float* __restrict__ b, long long n)
{
    const long long i = (long long)blockIdx.x * blockDim.x + threadIdx.x;
    if (i < n) a[i] += b[i];
}

// ---------------------------------------------------------------------------
// Host-side parameter unflattening (JAX pytrees flatten dict keys SORTED).
// ---------------------------------------------------------------------------
namespace {
struct Mlp { const float *w1,*b1,*w2,*b2,*lnb,*lng; };
struct Ffn { const float* bondW; Mlp gate, inter; const float* nodeW; };
struct EdgeP {
    Ffn fl, fr;
    const float *lnb,*lng;
    const float *mlW,*mlb,*mrW,*mrb,*nlW,*nlb,*nrW,*nrb;
    Mlp out;
    const float *sW,*sb;
};
struct NodeP {
    const float *cW,*cb;
    Mlp enet, gate;
    const float *lnb,*lng;
    const float *mW,*mb;
    Mlp nnet, out;
};
struct Cur { void* const* d; int i; const float* f() { return (const float*)d[i++]; } };
// mlp keys sorted: l1(W,b), l2(W,b), ln_b, ln_g
static Mlp rd_mlp(Cur& c){ Mlp m; m.w1=c.f(); m.b1=c.f(); m.w2=c.f(); m.b2=c.f(); m.lnb=c.f(); m.lng=c.f(); return m; }
// bond_ffn keys sorted: bond_W, gate, inter, node_W
static Ffn rd_ffn(Cur& c){ Ffn f; f.bondW=c.f(); f.gate=rd_mlp(c); f.inter=rd_mlp(c); f.nodeW=c.f(); return f; }
// edge block keys sorted: ffn_left, ffn_right, ln_b, ln_g, msg_l, msg_r, node_l, node_r, out, self
static EdgeP rd_edge(Cur& c){
    EdgeP e; e.fl=rd_ffn(c); e.fr=rd_ffn(c); e.lnb=c.f(); e.lng=c.f();
    e.mlW=c.f(); e.mlb=c.f(); e.mrW=c.f(); e.mrb=c.f();
    e.nlW=c.f(); e.nlb=c.f(); e.nrW=c.f(); e.nrb=c.f();
    e.out=rd_mlp(c); e.sW=c.f(); e.sb=c.f(); return e;
}
// node block keys sorted: centroid, edge_net, gate, ln_b, ln_g, msg_net, node_net, out
static NodeP rd_node(Cur& c){
    NodeP n; n.cW=c.f(); n.cb=c.f(); n.enet=rd_mlp(c); n.gate=rd_mlp(c);
    n.lnb=c.f(); n.lng=c.f(); n.mW=c.f(); n.mb=c.f();
    n.nnet=rd_mlp(c); n.out=rd_mlp(c); return n;
}
} // namespace

extern "C" void kernel_launch(void* const* d_in, const int* in_sizes, int n_in,
                              void* d_out, int out_size, void* d_ws, size_t ws_size,
                              hipStream_t stream)
{
    (void)in_sizes; (void)n_in; (void)out_size; (void)ws_size;
    const float* X   = (const float*)d_in[0];   // (N,256)
    const float* EA  = (const float*)d_in[1];   // (E,128)
    const float* NEx = (const float*)d_in[2];   // (N,8)
    const float* BEx = (const float*)d_in[3];   // (E,8)
    const int*   EI  = (const int*)d_in[4];     // (2,E)
    const int* rowi = EI;            // edge_index[0] = row / left
    const int* coli = EI + NEDGE;    // edge_index[1] = col / right

    Cur c{d_in, 5};
    EdgeP ep[2]; NodeP np[2];
    for (int b = 0; b < 2; ++b) { ep[b] = rd_edge(c); np[b] = rd_node(c); }  // "edge" < "node"

    float* ws = (float*)d_ws;
    size_t off = 0;
    auto alloc = [&](size_t n){ float* p = ws + off; off += (n + 63) & ~(size_t)63; return p; };
    float* hn  = alloc((size_t)NNODE * 256);
    float* he  = alloc((size_t)NEDGE * 128);
    float* et1 = alloc((size_t)NEDGE * 256);
    float* et2 = alloc((size_t)NEDGE * 256);
    float* nt1 = alloc((size_t)NNODE * 256);
    float* nt2 = alloc((size_t)NNODE * 256);
    float* na  = alloc((size_t)NNODE * 256);
    float* ef1 = alloc((size_t)NEDGE * 128);
    float* eg  = alloc((size_t)NEDGE * 32);
    float* eu  = alloc((size_t)NEDGE * 128);
    float* ns  = alloc((size_t)NNODE * 128);
    float* eo  = et2;  // reuse (free at that point)

    auto gemm = [&](const float* A, int lda, const int* g,
                    const float* W, int ldw, int wr0, const float* bias,
                    float* C, int ldc, int M, int N, int K, int beta, int epi){
        dim3 grid((unsigned)((N + 63) / 64), (unsigned)((M + 127) / 128));
        gemm_bf16_wmma<<<grid, 256, 0, stream>>>(A, lda, g, W, ldw, wr0, bias,
                                                 C, ldc, M, N, K, beta, epi);
    };
    auto ln = [&](const float* Xp, const float* Rp, const float* g, const float* b,
                  float* Y, int rows, int D, int relu){
        ln_kernel<<<rows, D, 0, stream>>>(Xp, Rp, g, b, Y, D, relu);
    };
    auto zero = [&](float* p, long long n){
        fill_zero<<<(unsigned)((n + 255) / 256), 256, 0, stream>>>(p, n);
    };

    const float* cx = X;   // current node features
    const float* ce = EA;  // current edge features
    float* nodeF = (float*)d_out;
    float* edgeF = (float*)d_out + (size_t)NNODE * 256;

    for (int b = 0; b < 2; ++b) {
        // ================= node block =================
        const NodeP& P = np[b];
        float* nout = (b == 1) ? nodeF : hn;

        // h_node = mlp(node_net, x)
        gemm(cx, 256, nullptr, P.nnet.w1, 256, 0, P.nnet.b1, nt1, 256, NNODE, 256, 256, 0, 0);
        ln(nt1, nullptr, P.nnet.lng, P.nnet.lnb, nt1, NNODE, 256, 1);
        gemm(nt1, 256, nullptr, P.nnet.w2, 256, 0, P.nnet.b2, nt2, 256, NNODE, 256, 256, 0, 0);
        // h_edge = mlp(edge_net, edge_attr)
        gemm(ce, 128, nullptr, P.enet.w1, 256, 0, P.enet.b1, et1, 256, NEDGE, 256, 128, 0, 0);
        ln(et1, nullptr, P.enet.lng, P.enet.lnb, et1, NEDGE, 256, 1);
        gemm(et1, 256, nullptr, P.enet.w2, 256, 0, P.enet.b2, et2, 256, NEDGE, 256, 256, 0, 0);
        // s = h_edge + h_node[col] + h_node[row];  msg = s @ Wm + bm
        add3_gather<<<NEDGE, 256, 0, stream>>>(et2, nt2, coli, rowi, et1, 256);
        gemm(et1, 256, nullptr, P.mW, 256, 0, P.mb, et2, 256, NEDGE, 256, 256, 0, 0);
        // gate hidden: concat segments as accumulating GEMMs (W rows 0/128/384/392/648)
        gemm(ce,  128, nullptr, P.gate.w1, 256, 0,   P.gate.b1, et1, 256, NEDGE, 256, 128, 0, 0);
        gemm(cx,  256, coli,    P.gate.w1, 256, 128, nullptr,   et1, 256, NEDGE, 256, 256, 1, 0);
        gemm(NEx, 8,   coli,    P.gate.w1, 256, 384, nullptr,   et1, 256, NEDGE, 256, 8,   1, 0);
        gemm(cx,  256, rowi,    P.gate.w1, 256, 392, nullptr,   et1, 256, NEDGE, 256, 256, 1, 0);
        gemm(NEx, 8,   rowi,    P.gate.w1, 256, 648, nullptr,   et1, 256, NEDGE, 256, 8,   1, 0);
        ln(et1, nullptr, P.gate.lng, P.gate.lnb, et1, NEDGE, 256, 1);
        // msg *= sigmoid(gate_l2)  -- fused epilogue
        gemm(et1, 256, nullptr, P.gate.w2, 256, 0, P.gate.b2, et2, 256, NEDGE, 256, 256, 0, 1);
        // aggr = segment_sum(msg, row)
        zero(na, (long long)NNODE * 256);
        scatter_add<<<NEDGE, 256, 0, stream>>>(et2, rowi, na, 256);
        // out = mlp(out_net, centroid(x) + aggr);  LN(out + x)
        gemm(cx, 256, nullptr, P.cW, 256, 0, P.cb, nt1, 256, NNODE, 256, 256, 0, 0);
        add_inplace<<<(NNODE * 256 + 255) / 256, 256, 0, stream>>>(nt1, na, (long long)NNODE * 256);
        gemm(nt1, 256, nullptr, P.out.w1, 256, 0, P.out.b1, nt2, 256, NNODE, 256, 256, 0, 0);
        ln(nt2, nullptr, P.out.lng, P.out.lnb, nt2, NNODE, 256, 1);
        gemm(nt2, 256, nullptr, P.out.w2, 256, 0, P.out.b2, nt1, 256, NNODE, 256, 256, 0, 0);
        ln(nt1, cx, P.lng, P.lnb, nout, NNODE, 256, 0);

        // ================= edge block =================
        const float* hv = nout;           // updated node features
        const EdgeP& Q = ep[b];
        float* eout = (b == 1) ? edgeF : he;

        for (int side = 0; side < 2; ++side) {
            const Ffn& F  = side ? Q.fr : Q.fl;
            const int* gi = side ? coli : rowi;   // node_in / upd gather index
            const int* si = side ? rowi : coli;   // segment_sum scatter index
            // pre = h_bond @ bond_W + h_node[gi] @ node_W
            gemm(ce, 128, nullptr, F.bondW, 256, 0, nullptr, et1, 256, NEDGE, 256, 128, 0, 0);
            gemm(hv, 256, gi,      F.nodeW, 256, 0, nullptr, et1, 256, NEDGE, 256, 256, 1, 0);
            // inter = mlp(inter, pre)
            gemm(et1, 256, nullptr, F.inter.w1, 256, 0, F.inter.b1, et2, 256, NEDGE, 256, 256, 0, 0);
            ln(et2, nullptr, F.inter.lng, F.inter.lnb, et2, NEDGE, 256, 1);
            gemm(et2, 256, nullptr, F.inter.w2, 128, 0, F.inter.b2, ef1, 128, NEDGE, 128, 256, 0, 0);
            // gate = mlp(gate, concat[h_bond, h_node[gi], bond_extra])  (hidden 32)
            gemm(ce,  128, nullptr, F.gate.w1, 32, 0,   F.gate.b1, eg, 32, NEDGE, 32, 128, 0, 0);
            gemm(hv,  256, gi,      F.gate.w1, 32, 128, nullptr,   eg, 32, NEDGE, 32, 256, 1, 0);
            gemm(BEx, 8,   nullptr, F.gate.w1, 32, 384, nullptr,   eg, 32, NEDGE, 32, 8,   1, 0);
            ln(eg, nullptr, F.gate.lng, F.gate.lnb, eg, NEDGE, 32, 1);
            gemm(eg, 32, nullptr, F.gate.w2, 128, 0, F.gate.b2, ef1, 128, NEDGE, 128, 32, 0, 1); // *=sigmoid
            // m = segment_sum(m, si)[gi];  upd (+)= m @ msg_W + b
            zero(ns, (long long)NNODE * 128);
            scatter_add<<<NEDGE, 128, 0, stream>>>(ef1, si, ns, 128);
            gemm(ns, 128, gi, side ? Q.mrW : Q.mlW, 128, 0, side ? Q.mrb : Q.mlb,
                 eu, 128, NEDGE, 128, 128, side ? 1 : 0, 0);
        }
        // upd += node_l(h_node[left]) + node_r(h_node[right]) + self(h_bond)
        gemm(hv, 256, rowi,    Q.nlW, 128, 0, Q.nlb, eu, 128, NEDGE, 128, 256, 1, 0);
        gemm(hv, 256, coli,    Q.nrW, 128, 0, Q.nrb, eu, 128, NEDGE, 128, 256, 1, 0);
        gemm(ce, 128, nullptr, Q.sW,  128, 0, Q.sb,  eu, 128, NEDGE, 128, 128, 1, 0);
        // out = mlp(out, upd);  h_edge = LN(out + h_bond)
        gemm(eu, 128, nullptr, Q.out.w1, 128, 0, Q.out.b1, ef1, 128, NEDGE, 128, 128, 0, 0);
        ln(ef1, nullptr, Q.out.lng, Q.out.lnb, ef1, NEDGE, 128, 1);
        gemm(ef1, 128, nullptr, Q.out.w2, 128, 0, Q.out.b2, eo, 128, NEDGE, 128, 128, 0, 0);
        ln(eo, ce, Q.lng, Q.lnb, eout, NEDGE, 128, 0);

        cx = nout; ce = eout;
    }
}
